// SupConLoss_2516850835771
// MI455X (gfx1250) — matrix operand order repaired and checked by reading, not compiled, not run on hardware
//
#include <hip/hip_runtime.h>
#include <hip/hip_bf16.h>

typedef __attribute__((ext_vector_type(16))) __bf16 v16bf;
typedef __attribute__((ext_vector_type(8)))  float  v8f;

#define N_ROWS 8192
#define DIM    256
#define INV_T  (1.0f / 0.07f)
#define L2E    1.4426950408889634f
#define LN2    0.6931471805599453f
#define SCALE2 (INV_T * L2E)        /* logits tracked in base-2 units */

// ---------------------------------------------------------------------------
// Kernel 1: build bf16 transposed feature matrix FT[DIM][N_ROWS] and zero the
// global accumulators.
// ---------------------------------------------------------------------------
__global__ __launch_bounds__(256) void supcon_convert(
    const float* __restrict__ F, __bf16* __restrict__ FT, float* __restrict__ acc)
{
    int tid = blockIdx.x * 256 + threadIdx.x;
    if (tid == 0) { acc[0] = 0.0f; acc[1] = 0.0f; }
    int n = tid & (N_ROWS - 1);   // column of FT (contiguous -> coalesced store)
    int d = tid >> 13;            // row of FT
    FT[(size_t)d * N_ROWS + n] = (__bf16)F[(size_t)n * DIM + d];
}

// ---------------------------------------------------------------------------
// Kernel 2: fused sim-tile + online masked log-softmax statistics.
// One block (8 waves) per 32-row i-tile (two 16-row WMMA tiles sharing every
// B load); wave w handles j-tiles w, w+8, ...
// ---------------------------------------------------------------------------
__global__ __launch_bounds__(256) void supcon_main(
    const float* __restrict__ F, const int* __restrict__ labels,
    const __bf16* __restrict__ FT, float* __restrict__ acc)
{
    const int iBase = blockIdx.x * 32;
    const int wave  = threadIdx.x >> 5;   // 0..7
    const int lane  = threadIdx.x & 31;
    const int ln16  = lane & 15;
    const int lh    = lane >> 4;          // 0 or 1

    // ---- A operands: two 16-row tiles, K=0..255, ISA A-matrix layout.
    // lanes 0-15: elems 0..7 = K kc*32+0..7,  elems 8..15 = K kc*32+16..23
    // lanes16-31: elems 0..7 = K kc*32+8..15, elems 8..15 = K kc*32+24..31
    v16bf a0[8], a1[8];
    const float* frow0 = F + (size_t)(iBase + ln16) * DIM;
    const float* frow1 = frow0 + (size_t)16 * DIM;
    #pragma unroll
    for (int kc = 0; kc < 8; ++kc) {
        const int k0 = kc * 32 + lh * 8;
        #pragma unroll
        for (int e = 0; e < 8; ++e) {
            a0[kc][e]     = (__bf16)frow0[k0 + e];
            a0[kc][8 + e] = (__bf16)frow0[k0 + 16 + e];
            a1[kc][e]     = (__bf16)frow1[k0 + e];
            a1[kc][8 + e] = (__bf16)frow1[k0 + 16 + e];
        }
    }

    // Labels of the C-tile rows this lane owns (M = t*16 + lh*8 + v).
    int labI[2][8];
    #pragma unroll
    for (int v = 0; v < 8; ++v) {
        labI[0][v] = labels[iBase      + lh * 8 + v];
        labI[1][v] = labels[iBase + 16 + lh * 8 + v];
    }

    // Per-(row, lane) online softmax state (base-2 units) + masked sums.
    float mrow[2][8], srow[2][8], Prow[2][8], Crow[2][8];
    #pragma unroll
    for (int t = 0; t < 2; ++t)
        #pragma unroll
        for (int v = 0; v < 8; ++v) {
            mrow[t][v] = -3.4e38f; srow[t][v] = 0.0f;
            Prow[t][v] = 0.0f;     Crow[t][v] = 0.0f;
        }

    const int krowBase = ln16 + 16 * lh;  // unused; this lane's B operand is column N = ln16

    for (int jt = wave; jt < N_ROWS / 16; jt += 8) {
        const int jBase = jt * 16;
        const int labJ  = labels[jBase + ln16];

        v8f c0 = {}, c1 = {};
        #pragma unroll
        for (int kc = 0; kc < 8; ++kc) {
            // B operand: column N = ln16, K in ISA order (lh*8+e, 16+lh*8+e)
            v16bf b;   // one gather feeds BOTH accumulators
            #pragma unroll
            for (int e = 0; e < 8; ++e) {
                b[e]     = FT[(size_t)(kc * 32 + lh * 8 + e) * N_ROWS + jBase + ln16];
                b[8 + e] = FT[(size_t)(kc * 32 + 16 + lh * 8 + e) * N_ROWS + jBase + ln16];
            }
            c0 = __builtin_amdgcn_wmma_f32_16x16x32_bf16(
                     false, a0[kc], false, b, (short)0, c0, false, false);
            c1 = __builtin_amdgcn_wmma_f32_16x16x32_bf16(
                     false, a1[kc], false, b, (short)0, c1, false, false);
        }

        // Epilogue: element (M = t*16 + lh*8 + v, N = ln16) of the sim tiles.
        #pragma unroll
        for (int t = 0; t < 2; ++t) {
            #pragma unroll
            for (int v = 0; v < 8; ++v) {
                const float x2   = (t ? c1[v] : c0[v]) * SCALE2;   // base-2 logit
                const float mOld = mrow[t][v];
                const float mNew = fmaxf(mOld, x2);
                srow[t][v] = srow[t][v] * __builtin_amdgcn_exp2f(mOld - mNew)
                           +              __builtin_amdgcn_exp2f(x2   - mNew);
                mrow[t][v] = mNew;
                const int gi = iBase + t * 16 + lh * 8 + v;
                const int gj = jBase + ln16;
                if ((labI[t][v] == labJ) && (gi != gj)) { Prow[t][v] += x2; Crow[t][v] += 1.0f; }
            }
        }
    }

    // ---- Butterfly reduce over the 16 lanes sharing the same rows.
    // (xor 16 would mix different M rows -> only xor 1,2,4,8)
    #pragma unroll
    for (int off = 1; off < 16; off <<= 1) {
        #pragma unroll
        for (int t = 0; t < 2; ++t)
            #pragma unroll
            for (int v = 0; v < 8; ++v) {
                const float m2 = __shfl_xor(mrow[t][v], off, 32);
                const float s2 = __shfl_xor(srow[t][v], off, 32);
                const float mN = fmaxf(mrow[t][v], m2);
                srow[t][v] = srow[t][v] * __builtin_amdgcn_exp2f(mrow[t][v] - mN)
                           + s2         * __builtin_amdgcn_exp2f(m2         - mN);
                mrow[t][v] = mN;
                Prow[t][v] += __shfl_xor(Prow[t][v], off, 32);
                Crow[t][v] += __shfl_xor(Crow[t][v], off, 32);
            }
    }

    // ---- Merge the 8 wave-partials through LDS (32 rows per block).
    __shared__ float ls[4][8][32];   // [stat][wave][row]
    if (ln16 == 0) {
        #pragma unroll
        for (int t = 0; t < 2; ++t)
            #pragma unroll
            for (int v = 0; v < 8; ++v) {
                const int r = t * 16 + lh * 8 + v;
                ls[0][wave][r] = mrow[t][v];
                ls[1][wave][r] = srow[t][v];
                ls[2][wave][r] = Prow[t][v];
                ls[3][wave][r] = Crow[t][v];
            }
    }
    __syncthreads();

    if (threadIdx.x < 32) {
        const int r = threadIdx.x;
        float m = -3.4e38f, s = 0.0f, P = 0.0f, C = 0.0f;
        #pragma unroll
        for (int w = 0; w < 8; ++w) {
            const float m2 = ls[0][w][r], s2 = ls[1][w][r];
            const float mN = fmaxf(m, m2);
            s = s * __builtin_amdgcn_exp2f(m - mN) + s2 * __builtin_amdgcn_exp2f(m2 - mN);
            m = mN;
            P += ls[2][w][r];
            C += ls[3][w][r];
        }
        float contrib = 0.0f, cnt = 0.0f;
        if (C > 0.0f) {
            // convert base-2 stats to natural units:
            //   sum_mask(log_prob) = (P - C*m)*ln2 - C*log(s + 1e-9)   [s is 2^-m scaled]
            contrib = -((P - C * m) * LN2 - C * logf(s + 1e-9f)) / C;
            cnt = 1.0f;
        }
        atomicAdd(&acc[0], contrib);
        atomicAdd(&acc[1], cnt);
    }
}

// ---------------------------------------------------------------------------
// Kernel 3: finalize scalar.
// ---------------------------------------------------------------------------
__global__ void supcon_finalize(const float* __restrict__ acc, float* __restrict__ out)
{
    const float t = acc[0], n = acc[1];
    out[0] = (n > 0.0f) ? (t / fmaxf(n, 1.0f)) : 0.0f;
}

extern "C" void kernel_launch(void* const* d_in, const int* in_sizes, int n_in,
                              void* d_out, int out_size, void* d_ws, size_t ws_size,
                              hipStream_t stream)
{
    const float* F      = (const float*)d_in[0];
    const int*   labels = (const int*)d_in[1];
    float*       out    = (float*)d_out;

    __bf16* FT  = (__bf16*)d_ws;                                    // 4 MB
    float*  acc = (float*)((char*)d_ws + (size_t)DIM * N_ROWS * 2); // 8 B

    supcon_convert<<<(N_ROWS * DIM) / 256, 256, 0, stream>>>(F, FT, acc);
    supcon_main<<<N_ROWS / 32, 256, 0, stream>>>(F, labels, FT, acc);
    supcon_finalize<<<1, 1, 0, stream>>>(acc, out);
}
